// DeformableAttention3D_75093208203557
// MI455X (gfx1250) — matrix operand order repaired and checked by reading, not compiled, not run on hardware
//
#include <hip/hip_runtime.h>
#include <hip/hip_bf16.h>
#include <math.h>

typedef __attribute__((ext_vector_type(16))) _Float16 v16h;
typedef __attribute__((ext_vector_type(8)))  float    v8f;

// ---------------- WMMA fragment helpers (CDNA5 16x16x32 f16, wave32) ----------
// A (16x32, MxK): lane l: m = l&15; kh = l>>4; elem e -> K = (e<8?0:16)+kh*8+(e&7)
// B (32x16, KxN): lane l: n = l&15; elem e -> K = (e<8?0:16)+(l>>4)*8+(e&7)
// C/D (16x16 f32): lane l: n = l&15; vgpr r -> m = r + 8*(l>>4)

__device__ __forceinline__ v16h load_a_f32(const float* __restrict__ A, int lda,
                                           int row0, int k0, int lane, float scale) {
  int m = lane & 15, kh = lane >> 4;
  v16h a;
#pragma unroll
  for (int e = 0; e < 16; ++e) {
    int k = ((e < 8) ? 0 : 16) + kh * 8 + (e & 7);
    a[e] = (_Float16)(A[(size_t)(row0 + m) * lda + k0 + k] * scale);
  }
  return a;
}

__device__ __forceinline__ v16h load_b_f32(const float* __restrict__ B, int ldb,
                                           int k0, int col0, int lane) {
  int n = lane & 15, kh = lane >> 4;
  v16h bb;
#pragma unroll
  for (int e = 0; e < 16; ++e) {
    int k = ((e < 8) ? 0 : 16) + kh * 8 + (e & 7);
    bb[e] = (_Float16)B[(size_t)(k0 + k) * ldb + col0 + n];
  }
  return bb;
}

__device__ __forceinline__ v16h load_a_f16(const _Float16* __restrict__ A, int lda,
                                           int row0, int k0, int lane) {
  int m = lane & 15, kh = lane >> 4;
  v16h a;
#pragma unroll
  for (int e = 0; e < 16; ++e) {
    int k = ((e < 8) ? 0 : 16) + kh * 8 + (e & 7);
    a[e] = A[(row0 + m) * lda + k0 + k];
  }
  return a;
}

// B[k][n] sourced from P[col0+n][k0+k] (i.e. row-major P, used transposed)
__device__ __forceinline__ v16h load_b_f16t(const _Float16* __restrict__ P, int ldp,
                                            int k0, int col0, int lane) {
  int n = lane & 15, kh = lane >> 4;
  v16h bb;
#pragma unroll
  for (int e = 0; e < 16; ++e) {
    int k = ((e < 8) ? 0 : 16) + kh * 8 + (e & 7);
    bb[e] = P[(col0 + n) * ldp + k0 + k];
  }
  return bb;
}

// ---------------- Kernel 1: q = wq @ x (grouped), q[bg][o][p] ------------------
__global__ __launch_bounds__(256) void qproj_kernel(const float* __restrict__ x,
                                                    const float* __restrict__ wq,
                                                    float* __restrict__ qbuf) {
  int lane = threadIdx.x & 31, wave = threadIdx.x >> 5;
  int job = blockIdx.x * 8 + wave;          // 4096 jobs = 8 bg * 8 mt * 64 nt
  int bg  = job >> 9;
  int rem = job & 511;
  int mt  = rem >> 6;
  int nt  = rem & 63;
  int b = bg >> 2, g = bg & 3;
  const float* W = wq + g * 128 * 64;                       // 128x64 row-major
  const float* X = x + ((size_t)b * 256 + g * 64) * 1024;   // 64x1024
  v8f c = {};
#pragma unroll
  for (int k0 = 0; k0 < 64; k0 += 32) {
    v16h a  = load_a_f32(W, 64, mt * 16, k0, lane, 1.0f);
    v16h bf = load_b_f32(X, 1024, k0, nt * 16, lane);
    c = __builtin_amdgcn_wmma_f32_16x16x32_f16(false, a, false, bf, (short)0, c, false, false);
  }
  float* C = qbuf + (size_t)bg * 128 * 1024;
  int n = lane & 15, mh = lane >> 4;
#pragma unroll
  for (int r = 0; r < 8; ++r)
    C[(size_t)(mt * 16 + mh * 8 + r) * 1024 + nt * 16 + n] = c[r];
}

// ---------------- Kernel 2: offsets -> normalized sample grid ------------------
__global__ __launch_bounds__(128) void offset_kernel(const float* __restrict__ qbuf,
                                                     const float* __restrict__ dw,
                                                     const float* __restrict__ dwb,
                                                     const float* __restrict__ pw,
                                                     float* __restrict__ gridkv) {
  int bg = blockIdx.x >> 7;
  int pj = blockIdx.x & 127;                 // pj = od*64 + oh*8 + ow
  int od = pj >> 6, oh = (pj >> 3) & 7, ow = pj & 7;
  int c  = threadIdx.x;                      // channel 0..127
  const float* qc = qbuf + ((size_t)bg * 128 + c) * 1024;
  float acc = dwb[c];
  for (int kd = 0; kd < 4; ++kd) {
    int id = od * 2 - 1 + kd;
    if (id < 0 || id >= 4) continue;
    for (int kh = 0; kh < 4; ++kh) {
      int ih = oh * 2 - 1 + kh;
      if (ih < 0 || ih >= 16) continue;
      for (int kw = 0; kw < 4; ++kw) {
        int iw = ow * 2 - 1 + kw;
        if (iw < 0 || iw >= 16) continue;
        acc += qc[id * 256 + ih * 16 + iw] * dw[c * 64 + kd * 16 + kh * 4 + kw];
      }
    }
  }
  // exact GELU
  float ge = 0.5f * acc * (1.0f + erff(acc * 0.70710678118654752f));
  __shared__ float sh[128];
  sh[c] = ge;
  __syncthreads();
  if (c < 3) {
    float o = 0.f;
    for (int i = 0; i < 128; ++i) o += pw[c * 128 + i] * sh[i];
    float offv  = tanhf(o) * 2.0f;
    float basec = (c == 0) ? (float)od : (c == 1) ? (float)oh : (float)ow;
    float denom = (c == 0) ? 1.0f : 7.0f;    // max(fd-1,1)=1, hd-1=wd-1=7
    gridkv[((size_t)bg * 128 + pj) * 3 + c] = 2.0f * (basec + offv) / denom - 1.0f;
  }
}

// ---------------- Kernel 3a: trilinear gather kv[bg][c][pj] --------------------
__global__ __launch_bounds__(64) void sample_kernel(const float* __restrict__ x,
                                                    const float* __restrict__ gridkv,
                                                    float* __restrict__ kvbuf) {
  int bg = blockIdx.x >> 7;
  int pj = blockIdx.x & 127;
  int c  = threadIdx.x;                      // 0..63
  int b = bg >> 2, g = bg & 3;
  const float* g3 = gridkv + ((size_t)bg * 128 + pj) * 3;
  // faithful to reference: comp0->x(W=16), comp1->y(H=16), comp2->z(D=4)
  float ix = ((g3[0] + 1.f) * 16.f - 1.f) * 0.5f;
  float iy = ((g3[1] + 1.f) * 16.f - 1.f) * 0.5f;
  float iz = ((g3[2] + 1.f) * 4.f  - 1.f) * 0.5f;
  float x0 = floorf(ix), y0 = floorf(iy), z0 = floorf(iz);
  float tx = ix - x0, ty = iy - y0, tz = iz - z0;
  const float* vol = x + ((size_t)b * 256 + g * 64 + c) * 1024;
  float out = 0.f;
#pragma unroll
  for (int dz = 0; dz < 2; ++dz)
#pragma unroll
    for (int dy = 0; dy < 2; ++dy)
#pragma unroll
      for (int dx = 0; dx < 2; ++dx) {
        float xc = x0 + dx, yc = y0 + dy, zc = z0 + dz;
        float wgt = (dx ? tx : 1.f - tx) * (dy ? ty : 1.f - ty) * (dz ? tz : 1.f - tz);
        if (xc >= 0.f && xc < 16.f && yc >= 0.f && yc < 16.f && zc >= 0.f && zc < 4.f) {
          int xi = (int)xc, yi = (int)yc, zi = (int)zc;
          out += vol[(zi * 16 + yi) * 16 + xi] * wgt;
        }
      }
  kvbuf[((size_t)bg * 64 + c) * 128 + pj] = out;
}

// ---------------- Kernel 3b: k,v = wk/wv @ kv (small, VALU) --------------------
__global__ __launch_bounds__(128) void kvproj_kernel(const float* __restrict__ kvbuf,
                                                     const float* __restrict__ wk,
                                                     const float* __restrict__ wv,
                                                     float* __restrict__ kbuf,
                                                     float* __restrict__ vbuf) {
  int bg = blockIdx.x >> 7;
  int o  = blockIdx.x & 127;
  int pj = threadIdx.x;
  int g  = bg & 3;
  const float* kvb = kvbuf + (size_t)bg * 64 * 128 + pj;
  const float* wkr = wk + (g * 128 + o) * 64;
  const float* wvr = wv + (g * 128 + o) * 64;
  float ak = 0.f, av = 0.f;
#pragma unroll
  for (int i = 0; i < 64; ++i) {
    float kvv = kvb[i * 128];
    ak += kvv * wkr[i];
    av += kvv * wvr[i];
  }
  kbuf[((size_t)bg * 128 + o) * 128 + pj] = ak;
  vbuf[((size_t)bg * 128 + o) * 128 + pj] = av;
}

// ---------------- Kernel 4: fused sim + CPB bias MLP + softmax + attn@v -------
__global__ __launch_bounds__(256) void attn_kernel(
    const float* __restrict__ qbuf, const float* __restrict__ kbuf,
    const float* __restrict__ vbuf, const float* __restrict__ gridkv,
    const float* __restrict__ w0, const float* __restrict__ b0,
    const float* __restrict__ w1, const float* __restrict__ b1,
    const float* __restrict__ w2, const float* __restrict__ b2,
    float* __restrict__ aobuf) {
  __shared__ float    S[2 * 16 * 128];      // sim+bias, f32  (16 KB)
  __shared__ _Float16 Hbuf[256 * 64];       // MLP hidden chunks (32 KB)
  __shared__ _Float16 Pbuf[2 * 16 * 128];   // softmax probs, f16 (8 KB)
  __shared__ float    gkv[128 * 3];         // sample grid (1.5 KB)
  __shared__ float    red[256];             // softmax reductions (1 KB)
  __shared__ float    w0s[192], b0s[64], w2s[128], b2s[2];

  int tid = threadIdx.x;
  int lane = tid & 31, wave = tid >> 5;     // 8 waves
  int blk = blockIdx.x;                     // 512 = 8 bg * 64 qtiles
  int bg = blk >> 6;
  int qt = blk & 63;
  int qbase = qt * 16;
  int b = bg >> 2, g = bg & 3;

  for (int i = tid; i < 128 * 3; i += 256) gkv[i] = gridkv[(size_t)bg * 128 * 3 + i];
  if (tid < 192) w0s[tid] = w0[tid];
  if (tid < 64)  b0s[tid] = b0[tid];
  if (tid < 128) w2s[tid] = w2[tid];
  if (tid < 2)   b2s[tid] = b2[tid];

  const float* Q = qbuf + (size_t)bg * 128 * 1024;
  const float* K = kbuf + (size_t)bg * 128 * 128;
  const float* V = vbuf + (size_t)bg * 128 * 128;
  const float scale = 0.125f;               // 64^-0.5

  // ---- sim: 16 tile jobs (2 heads x 8 j-tiles), 2 per wave ----
#pragma unroll
  for (int jj = 0; jj < 2; ++jj) {
    int job = wave + jj * 8;
    int hsub = job >> 3, jt = job & 7;
    v8f c = {};
#pragma unroll
    for (int k0 = 0; k0 < 64; k0 += 32) {
      // A[i][d] = scale * Q[hsub*64+d][qbase+i] (transposed gather)
      int m = lane & 15, kh = lane >> 4;
      v16h a;
#pragma unroll
      for (int e = 0; e < 16; ++e) {
        int k = ((e < 8) ? 0 : 16) + kh * 8 + (e & 7);
        a[e] = (_Float16)(Q[(size_t)(hsub * 64 + k0 + k) * 1024 + qbase + m] * scale);
      }
      v16h bf = load_b_f32(K + (size_t)hsub * 64 * 128, 128, k0, jt * 16, lane);
      c = __builtin_amdgcn_wmma_f32_16x16x32_f16(false, a, false, bf, (short)0, c, false, false);
    }
    int n = lane & 15, mh = lane >> 4;
#pragma unroll
    for (int r = 0; r < 8; ++r)
      S[(hsub * 16 + mh * 8 + r) * 128 + jt * 16 + n] = c[r];
  }
  __syncthreads();

  // ---- hoist W1 B-fragments + b1 lane values into registers (reused 8x) ----
  v16h w1frag[2][4];
#pragma unroll
  for (int nt = 0; nt < 4; ++nt) {
    w1frag[0][nt] = load_b_f32(w1, 64, 0,  nt * 16, lane);
    w1frag[1][nt] = load_b_f32(w1, 64, 32, nt * 16, lane);
  }
  float b1v[4];
#pragma unroll
  for (int nt = 0; nt < 4; ++nt) b1v[nt] = b1[nt * 16 + (lane & 15)];

  // ---- CPB bias MLP: 2048 pairs in 8 chunks of 256 rows ----
  for (int chunk = 0; chunk < 8; ++chunk) {
    int row = chunk * 256 + tid;            // pair index: i*128 + j
    int i = row >> 7, j = row & 127;
    int qidx = qbase + i;
    float gq0 = 2.f * (float)(qidx >> 8) / 3.f - 1.f;
    float gq1 = 2.f * (float)((qidx >> 4) & 15) / 15.f - 1.f;
    float gq2 = 2.f * (float)(qidx & 15) / 15.f - 1.f;
    float p0 = gq0 - gkv[j * 3 + 0];
    float p1 = gq1 - gkv[j * 3 + 1];
    float p2 = gq2 - gkv[j * 3 + 2];
    float s0 = (p0 > 0.f ? 1.f : (p0 < 0.f ? -1.f : 0.f)) * log1pf(fabsf(p0));
    float s1 = (p1 > 0.f ? 1.f : (p1 < 0.f ? -1.f : 0.f)) * log1pf(fabsf(p1));
    float s2 = (p2 > 0.f ? 1.f : (p2 < 0.f ? -1.f : 0.f)) * log1pf(fabsf(p2));
    // layer 1 (K=3, VALU) -> Hbuf f16 row-major 256x64
#pragma unroll
    for (int o = 0; o < 64; ++o) {
      float t = s0 * w0s[o] + s1 * w0s[64 + o] + s2 * w0s[128 + o] + b0s[o];
      Hbuf[tid * 64 + o] = (_Float16)fmaxf(t, 0.f);
    }
    __syncthreads();
    // layer 2 via WMMA: (256x64)@(64x64) = 64 tile jobs, 8 per wave
    v8f frag[8];
#pragma unroll
    for (int jb = 0; jb < 8; ++jb) {
      int job = wave * 8 + jb;
      int rt = job >> 2, nt = job & 3;
      v8f c = {};
      v16h a0 = load_a_f16(Hbuf, 64, rt * 16, 0, lane);
      c = __builtin_amdgcn_wmma_f32_16x16x32_f16(false, a0, false, w1frag[0][nt], (short)0, c, false, false);
      v16h a1 = load_a_f16(Hbuf, 64, rt * 16, 32, lane);
      c = __builtin_amdgcn_wmma_f32_16x16x32_f16(false, a1, false, w1frag[1][nt], (short)0, c, false, false);
      frag[jb] = c;
    }
    __syncthreads();                         // all reads of Hbuf (h1) done
#pragma unroll
    for (int jb = 0; jb < 8; ++jb) {
      int job = wave * 8 + jb;
      int rt = job >> 2, nt = job & 3;
      int n = lane & 15, mh = lane >> 4;
#pragma unroll
      for (int r = 0; r < 8; ++r) {
        int m = mh * 8 + r;
        float v = frag[jb][r] + b1v[nt];
        Hbuf[(rt * 16 + m) * 64 + nt * 16 + n] = (_Float16)fmaxf(v, 0.f);
      }
    }
    __syncthreads();
    // layer 3 (64->2, VALU) + accumulate into S (one thread per pair)
    float o0 = b2s[0], o1 = b2s[1];
#pragma unroll
    for (int o = 0; o < 64; ++o) {
      float hh = (float)Hbuf[tid * 64 + o];
      o0 += hh * w2s[o * 2 + 0];
      o1 += hh * w2s[o * 2 + 1];
    }
    S[(0 * 16 + i) * 128 + j] += o0;
    S[(1 * 16 + i) * 128 + j] += o1;
    __syncthreads();
  }

  // ---- softmax over Pj=128: 32 rows, 8 threads per row ----
  {
    int rowid = tid >> 3;                   // 0..31 = hsub*16 + i
    int sub = tid & 7;
    float* row = &S[rowid * 128];
    float lmax = -1e30f;
#pragma unroll
    for (int t = 0; t < 16; ++t) lmax = fmaxf(lmax, row[sub + t * 8]);
    red[tid] = lmax;
    __syncthreads();
    float rmax = -1e30f;
#pragma unroll
    for (int u = 0; u < 8; ++u) rmax = fmaxf(rmax, red[(rowid << 3) + u]);
    float lsum = 0.f;
#pragma unroll
    for (int t = 0; t < 16; ++t) {
      int j = sub + t * 8;
      float e = expf(row[j] - rmax);
      row[j] = e;
      lsum += e;
    }
    __syncthreads();                         // red reads done before rewrite
    red[tid] = lsum;
    __syncthreads();
    float rsum = 0.f;
#pragma unroll
    for (int u = 0; u < 8; ++u) rsum += red[(rowid << 3) + u];
    float inv = 1.f / rsum;
#pragma unroll
    for (int t = 0; t < 16; ++t) {
      int j = sub + t * 8;
      Pbuf[rowid * 128 + j] = (_Float16)(row[j] * inv);
    }
  }
  __syncthreads();

  // ---- out = attn @ v^T : 8 jobs (2 heads x 4 d-tiles), one per wave ----
  {
    int hsub = wave >> 2, dt = wave & 3;
    v8f c = {};
#pragma unroll
    for (int k0 = 0; k0 < 128; k0 += 32) {
      v16h a  = load_a_f32(V + (size_t)(hsub * 64 + dt * 16) * 128, 128, 0, k0, lane, 1.0f);
      v16h bf = load_b_f16t(&Pbuf[hsub * 16 * 128], 128, k0, 0, lane);
      c = __builtin_amdgcn_wmma_f32_16x16x32_f16(false, a, false, bf, (short)0, c, false, false);
    }
    int h = g * 2 + hsub;
    int n = lane & 15, mh = lane >> 4;
    float* AO = aobuf + ((size_t)b * 512 + h * 64 + dt * 16) * 1024 + qbase;
#pragma unroll
    for (int r = 0; r < 8; ++r)
      AO[(size_t)(mh * 8 + r) * 1024 + n] = c[r];
  }
}

// ---------------- Kernel 5: out = wo @ attn_out + bo ---------------------------
__global__ __launch_bounds__(256) void oproj_kernel(const float* __restrict__ aobuf,
                                                    const float* __restrict__ wo,
                                                    const float* __restrict__ bo,
                                                    float* __restrict__ out) {
  int lane = threadIdx.x & 31, wave = threadIdx.x >> 5;
  int job = blockIdx.x * 8 + wave;          // 2048 jobs = 2 b * 16 mt * 64 nt
  int b   = job >> 10;
  int rem = job & 1023;
  int mt  = rem >> 6;
  int nt  = rem & 63;
  const float* X = aobuf + (size_t)b * 512 * 1024;
  v8f c = {};
#pragma unroll 4
  for (int k0 = 0; k0 < 512; k0 += 32) {
    v16h a  = load_a_f32(wo, 512, mt * 16, k0, lane, 1.0f);
    v16h bf = load_b_f32(X, 1024, k0, nt * 16, lane);
    c = __builtin_amdgcn_wmma_f32_16x16x32_f16(false, a, false, bf, (short)0, c, false, false);
  }
  int n = lane & 15, mh = lane >> 4;
  float* O = out + (size_t)b * 256 * 1024;
#pragma unroll
  for (int r = 0; r < 8; ++r) {
    int m = mt * 16 + mh * 8 + r;
    O[(size_t)m * 1024 + nt * 16 + n] = c[r] + bo[m];
  }
}

extern "C" void kernel_launch(void* const* d_in, const int* in_sizes, int n_in,
                              void* d_out, int out_size, void* d_ws, size_t ws_size,
                              hipStream_t stream) {
  const float* x   = (const float*)d_in[0];
  const float* wq  = (const float*)d_in[1];
  const float* wk  = (const float*)d_in[2];
  const float* wv  = (const float*)d_in[3];
  const float* dw  = (const float*)d_in[4];
  const float* dwb = (const float*)d_in[5];
  const float* pw  = (const float*)d_in[6];
  const float* w0  = (const float*)d_in[7];
  const float* b0  = (const float*)d_in[8];
  const float* w1  = (const float*)d_in[9];
  const float* b1  = (const float*)d_in[10];
  const float* w2  = (const float*)d_in[11];
  const float* b2  = (const float*)d_in[12];
  const float* wo  = (const float*)d_in[13];
  const float* bo  = (const float*)d_in[14];
  float* out = (float*)d_out;

  float* ws     = (float*)d_ws;
  float* qbuf   = ws;                        // 8*128*1024 = 1048576
  float* gridkv = qbuf + 1048576;            // 8*128*3    = 3072
  float* kvbuf  = gridkv + 3072;             // 8*64*128   = 65536
  float* kbuf   = kvbuf + 65536;             // 8*128*128  = 131072
  float* vbuf   = kbuf + 131072;             // 131072
  float* aobuf  = vbuf + 131072;             // 2*512*1024 = 1048576

  hipLaunchKernelGGL(qproj_kernel,  dim3(512),  dim3(256), 0, stream, x, wq, qbuf);
  hipLaunchKernelGGL(offset_kernel, dim3(1024), dim3(128), 0, stream, qbuf, dw, dwb, pw, gridkv);
  hipLaunchKernelGGL(sample_kernel, dim3(1024), dim3(64),  0, stream, x, gridkv, kvbuf);
  hipLaunchKernelGGL(kvproj_kernel, dim3(1024), dim3(128), 0, stream, kvbuf, wk, wv, kbuf, vbuf);
  hipLaunchKernelGGL(attn_kernel,   dim3(512),  dim3(256), 0, stream,
                     qbuf, kbuf, vbuf, gridkv, w0, b0, w1, b1, w2, b2, aobuf);
  hipLaunchKernelGGL(oproj_kernel,  dim3(256),  dim3(256), 0, stream, aobuf, wo, bo, out);
}
